// BiLSTM_34729105555633
// MI455X (gfx1250) — matrix operand order repaired and checked by reading, not compiled, not run on hardware
//
#include <hip/hip_runtime.h>
#include <hip/hip_bf16.h>
#include <stdint.h>

typedef __attribute__((ext_vector_type(16))) __bf16 v16bf;
typedef __attribute__((ext_vector_type(8)))  float  v8f;

#define B_   1024
#define T_   512
#define I_   46     // input features (padded to 64 for WMMA K; col 46 = bias)
#define H_   32
#define G_   128    // 4*H gate columns

#if __has_builtin(__builtin_amdgcn_tanhf)
__device__ __forceinline__ float tanh_fast(float x) { return __builtin_amdgcn_tanhf(x); }
#else
__device__ __forceinline__ float tanh_fast(float x) {
  return 2.0f * __builtin_amdgcn_rcpf(1.0f + __expf(-2.0f * x)) - 1.0f;
}
#endif
__device__ __forceinline__ float sig_fast(float x) { return 0.5f + 0.5f * tanh_fast(0.5f * x); }

// ---------------------------------------------------------------------------
// Forward LSTM scan: one wave32 per 16-row batch tile. bf16 WMMA, f32 accum.
//   gates(16x128) = x0(16x32)@Wih0 + x1(16x32)@Wih1 + h(16x32)@Whh    (C = 0)
// Bias is folded into W_ih column k=46 with x pad column 46 == 1.0.
// x time-slices are double-buffered in LDS via async global->LDS DMA.
// ---------------------------------------------------------------------------
__global__ __launch_bounds__(32)
void lstm_fwd_kernel(const float* __restrict__ x,
                     const float* __restrict__ Wih,
                     const float* __restrict__ Whh,
                     const float* __restrict__ bih,
                     const float* __restrict__ bhh,
                     float* __restrict__ hf)
{
  // Weights in B-operand order: [lane][16 bf16] = 32B per lane
  __shared__ __align__(32) __bf16 sWih[2][8][32][16];  // 16 KB
  __shared__ __align__(32) __bf16 sWhh[8][32][16];     //  8 KB
  __shared__ __align__(16) float  sx[2][16][48];       //  6 KB x double buffer
  __shared__ __align__(16) float  sh[16][32];          //  2 KB h transpose

  const int lane  = threadIdx.x;
  const int nl    = lane & 15;
  const int lhalf = lane >> 4;
  const int b0    = blockIdx.x * 16;
  const int kb0   = lhalf ? 8  : 0;    // A-layout K-group bases (16-bit data)
  const int kb1   = lhalf ? 24 : 16;

  // ---- stage W_ih (f32 -> bf16, K padded 46 -> 64, bias at k==46) ----------
  for (int c = 0; c < 2; ++c) {
    for (int j = 0; j < 8; ++j) {
      const int n  = j * 16 + nl;          // this lane's gate column
      const int k0 = c * 32 + 8 * lhalf;   // B-layout: this lane's K base (table order)
      #pragma unroll
      for (int q = 0; q < 16; ++q) {
        const int k = k0 + q + (q < 8 ? 0 : 8);
        float v = (k < I_) ? Wih[n * I_ + k]
                           : (k == I_ ? (bih[n] + bhh[n]) : 0.0f);
        sWih[c][j][lane][q] = (__bf16)v;
      }
    }
  }
  // ---- stage W_hh (single K-chunk of 32) -----------------------------------
  for (int j = 0; j < 8; ++j) {
    const int n = j * 16 + nl;
    #pragma unroll
    for (int q = 0; q < 16; ++q)
      sWhh[j][lane][q] = (__bf16)Whh[n * H_ + 8 * lhalf + q + (q < 8 ? 0 : 8)];
  }

  // ---- pad columns: col 46 = 1.0 (bias multiplier), col 47 = 0 -------------
  if (lane < 16) {
    sx[0][lane][46] = 1.0f; sx[0][lane][47] = 0.f;
    sx[1][lane][46] = 1.0f; sx[1][lane][47] = 0.f;
  }

  // ---- async prefetch of one x time-slice: 16 rows x 23 b64 = 368 DMAs -----
  // (x rows are only 8B-aligned -> b64 granularity). 368 = 11*32 + 16.
  auto prefetch_tile = [&](int t, int nbuf) {
    #pragma unroll
    for (int it = 0; it < 12; ++it) {
      if (it < 11 || lane < 16) {
        const int idx = lane + it * 32;
        const int row = idx / 23;
        const int col = (idx % 23) * 2;
        unsigned ldsa = (unsigned)(uintptr_t)&sx[nbuf][row][col];
        unsigned long long ga = (unsigned long long)(uintptr_t)
            (x + ((size_t)(b0 + row) * T_ + t) * I_ + col);
        asm volatile("global_load_async_to_lds_b64 %0, %1, off"
                     :: "v"(ldsa), "v"(ga) : "memory");
      }
    }
  };

  prefetch_tile(0, 0);

  v16bf hA;                                  // h in A-operand (bf16) layout
  #pragma unroll
  for (int q = 0; q < 16; ++q) hA[q] = (__bf16)0.0f;
  float cst[2][8];                           // cell state, C-layout
  #pragma unroll
  for (int p = 0; p < 2; ++p)
    #pragma unroll
    for (int r = 0; r < 8; ++r) cst[p][r] = 0.0f;
  v8f zc;                                    // shared zero C operand
  #pragma unroll
  for (int r = 0; r < 8; ++r) zc[r] = 0.0f;

  #pragma unroll 1
  for (int t = 0; t < T_; ++t) {
    const int buf = t & 1;
    // wait for this buffer's DMA; immediately launch next-step prefetch so the
    // DMA overlaps this step's WMMA + activation work.
    asm volatile("s_wait_asynccnt 0" ::: "memory");
    prefetch_tile((t + 1) & (T_ - 1), buf ^ 1);

    // build x A-operands for both K-chunks (rows: M = nl for every lane)
    const float* xrow = &sx[buf][nl][0];
    const float4 xv0 = *(const float4*)(xrow + kb0);
    const float4 xv1 = *(const float4*)(xrow + kb0 + 4);
    const float4 xv2 = *(const float4*)(xrow + kb1);
    const float4 xv3 = *(const float4*)(xrow + kb1 + 4);
    const float4 xv4 = *(const float4*)(xrow + 32 + kb0);
    const float4 xv5 = *(const float4*)(xrow + 32 + kb0 + 4);
    v16bf a0, a1;
    {
      const float f0[16] = {xv0.x,xv0.y,xv0.z,xv0.w, xv1.x,xv1.y,xv1.z,xv1.w,
                            xv2.x,xv2.y,xv2.z,xv2.w, xv3.x,xv3.y,xv3.z,xv3.w};
      const float f1[8]  = {xv4.x,xv4.y,xv4.z,xv4.w, xv5.x,xv5.y,xv5.z,xv5.w};
      #pragma unroll
      for (int q = 0; q < 16; ++q) a0[q] = (__bf16)f0[q];
      #pragma unroll
      for (int q = 0; q < 8; ++q)  { a1[q] = (__bf16)f1[q]; a1[8+q] = (__bf16)0.0f; }
    }

    // two halves of the hidden width to cap live accumulators at 4 tiles
    #pragma unroll
    for (int p = 0; p < 2; ++p) {
      v8f acc[4];
      #pragma unroll
      for (int gidx = 0; gidx < 4; ++gidx) {      // gidx: 0=i 1=f 2=g 3=o
        const int j = gidx * 2 + p;
        v16bf w0 = *(const v16bf*)&sWih[0][j][lane][0];
        v16bf w1 = *(const v16bf*)&sWih[1][j][lane][0];
        v16bf wh = *(const v16bf*)&sWhh[j][lane][0];
        v8f a;
        a = __builtin_amdgcn_wmma_f32_16x16x32_bf16(false, a0, false, w0,
                                                    (short)0, zc, false, false);
        a = __builtin_amdgcn_wmma_f32_16x16x32_bf16(false, a1, false, w1,
                                                    (short)0, a, false, false);
        a = __builtin_amdgcn_wmma_f32_16x16x32_bf16(false, hA, false, wh,
                                                    (short)0, a, false, false);
        acc[gidx] = a;
      }
      #pragma unroll
      for (int r = 0; r < 8; ++r) {
        const float iv = sig_fast(acc[0][r]);
        const float fv = sig_fast(acc[1][r]);
        const float gv = tanh_fast(acc[2][r]);
        const float ov = sig_fast(acc[3][r]);
        const float cn = fv * cst[p][r] + iv * gv;
        cst[p][r] = cn;
        sh[r + 8 * lhalf][p * 16 + nl] = ov * tanh_fast(cn);  // C-layout -> LDS
      }
    }

    // transpose h back into the bf16 A-operand layout (same-wave DS, in order)
    {
      const float4 h0 = *(const float4*)&sh[nl][kb0];
      const float4 h1 = *(const float4*)&sh[nl][kb0 + 4];
      const float4 h2 = *(const float4*)&sh[nl][kb1];
      const float4 h3 = *(const float4*)&sh[nl][kb1 + 4];
      const float hh[16] = {h0.x,h0.y,h0.z,h0.w, h1.x,h1.y,h1.z,h1.w,
                            h2.x,h2.y,h2.z,h2.w, h3.x,h3.y,h3.z,h3.w};
      #pragma unroll
      for (int q = 0; q < 16; ++q) hA[q] = (__bf16)hh[q];
    }
  }

  // ---- write final hidden state h_f (B x 32, f32) to workspace
  #pragma unroll
  for (int q = 0; q < 16; ++q)
    hf[(size_t)(b0 + nl) * H_ + 16 * lhalf + q] = sh[nl][16 * lhalf + q];
}

// ---------------------------------------------------------------------------
// Backward direction = ONE cell from zero state on x[:, T-1] (W_hh_b is dead,
// f-gate is dead since c0 = 0), then the tiny FC head.
// ---------------------------------------------------------------------------
__global__ __launch_bounds__(32)
void bwd_fc_kernel(const float* __restrict__ x,
                   const float* __restrict__ Wihb,
                   const float* __restrict__ bihb,
                   const float* __restrict__ bhhb,
                   const float* __restrict__ Wfc,
                   const float* __restrict__ bfc,
                   const float* __restrict__ hf,
                   float* __restrict__ out)
{
  __shared__ __align__(16) float sh[16][32];   // h_b tile

  const int lane  = threadIdx.x;
  const int nl    = lane & 15;
  const int lhalf = lane >> 4;
  const int b0    = blockIdx.x * 16;
  const int kb0   = lhalf ? 8  : 0;
  const int kb1   = lhalf ? 24 : 16;

  // A operand from x[:, T-1]
  const float* xr = x + ((size_t)(b0 + nl) * T_ + (T_ - 1)) * I_;
  v16bf a0, a1;
  #pragma unroll
  for (int q = 0; q < 8; ++q) {
    a0[q]     = (__bf16)xr[kb0 + q];
    a0[8 + q] = (__bf16)xr[kb1 + q];
    const int k = 32 + kb0 + q;
    a1[q]     = (k < I_) ? (__bf16)xr[k] : (__bf16)0.0f;
    a1[8 + q] = (__bf16)0.0f;
  }

  #pragma unroll
  for (int p = 0; p < 2; ++p) {
    const int jlist[3] = {p, 4 + p, 6 + p};    // i, g, o tiles (skip dead f)
    v8f acc[3];
    #pragma unroll
    for (int u = 0; u < 3; ++u) {
      const int j = jlist[u];
      const int n = j * 16 + nl;
      const float* wr = Wihb + n * I_;
      v16bf w0, w1;
      #pragma unroll
      for (int q = 0; q < 16; ++q) {
        w0[q] = (__bf16)wr[8 * lhalf + q + (q < 8 ? 0 : 8)];
        const int k = 32 + 8 * lhalf + q + (q < 8 ? 0 : 8);
        w1[q] = (k < I_) ? (__bf16)wr[k] : (__bf16)0.0f;
      }
      const float bb = bihb[n] + bhhb[n];
      v8f a;
      #pragma unroll
      for (int r = 0; r < 8; ++r) a[r] = bb;
      a = __builtin_amdgcn_wmma_f32_16x16x32_bf16(false, a0, false, w0,
                                                  (short)0, a, false, false);
      a = __builtin_amdgcn_wmma_f32_16x16x32_bf16(false, a1, false, w1,
                                                  (short)0, a, false, false);
      acc[u] = a;
    }
    #pragma unroll
    for (int r = 0; r < 8; ++r) {
      const float iv = sig_fast(acc[0][r]);
      const float gv = tanh_fast(acc[1][r]);
      const float ov = sig_fast(acc[2][r]);
      sh[r + 8 * lhalf][p * 16 + nl] = ov * tanh_fast(iv * gv);  // c0=0 -> c=i*g
    }
  }
  __syncthreads();

  // FC head: 16 rows x 8 outputs per block; lane -> (row = nl, 4 outputs)
  const int row   = nl;
  const int pbase = lhalf * 4;
  const float* hfr = hf + (size_t)(b0 + row) * H_;
  #pragma unroll
  for (int pp = 0; pp < 4; ++pp) {
    const int p = pbase + pp;
    float s = bfc[p];
    #pragma unroll 8
    for (int q = 0; q < 32; ++q) s += hfr[q]     * Wfc[p * 64 + q];
    #pragma unroll 8
    for (int q = 0; q < 32; ++q) s += sh[row][q] * Wfc[p * 64 + 32 + q];
    out[(size_t)(b0 + row) * 8 + p] = s;
  }
}

// ---------------------------------------------------------------------------
extern "C" void kernel_launch(void* const* d_in, const int* in_sizes, int n_in,
                              void* d_out, int out_size, void* d_ws, size_t ws_size,
                              hipStream_t stream) {
  (void)in_sizes; (void)n_in; (void)out_size; (void)ws_size;
  const float* x     = (const float*)d_in[0];
  const float* Wih_f = (const float*)d_in[1];
  const float* Whh_f = (const float*)d_in[2];
  const float* bih_f = (const float*)d_in[3];
  const float* bhh_f = (const float*)d_in[4];
  const float* Wih_b = (const float*)d_in[5];
  // d_in[6] (W_hh_b) is mathematically unused: backward runs one step from h0=0.
  const float* bih_b = (const float*)d_in[7];
  const float* bhh_b = (const float*)d_in[8];
  const float* Wfc   = (const float*)d_in[9];
  const float* bfc   = (const float*)d_in[10];

  float* hf = (float*)d_ws;   // 1024*32 f32 = 128 KB scratch for h_f

  lstm_fwd_kernel<<<B_ / 16, 32, 0, stream>>>(x, Wih_f, Whh_f, bih_f, bhh_f, hf);
  bwd_fc_kernel<<<B_ / 16, 32, 0, stream>>>(x, Wih_b, bih_b, bhh_b, Wfc, bfc,
                                            hf, (float*)d_out);
}
